// MoeFc_31275951850271
// MI455X (gfx1250) — matrix operand order repaired and checked by reading, not compiled, not run on hardware
//
#include <hip/hip_runtime.h>

// ---------------------------------------------------------------------------
// MoE FFN for MI455X (gfx1250, wave32, WMMA).
// Sparse top-2 expert evaluation; bf16 WMMA GEMMs with f32 accumulation.
// Weights are pre-transposed to [N][K] at conversion time so all LDS staging
// is vector b128 copies (via GLOBAL_LOAD_ASYNC_TO_LDS_B128 when available).
// ---------------------------------------------------------------------------

typedef __attribute__((ext_vector_type(16))) __bf16 v16bf;
typedef __attribute__((ext_vector_type(8)))  float  v8f;
typedef __attribute__((ext_vector_type(4)))  int    vi4;

#define TOK   4096   // B*S
#define DIM   1024   // D == O
#define NEXP  8

// ---- optional CDNA5 async global->LDS path (probed, with safe fallback) ----
#if defined(__has_builtin)
#if __has_builtin(__builtin_amdgcn_global_load_async_to_lds_b128)
#define HAS_ASYNC_LDS 1
#endif
#endif

#ifdef HAS_ASYNC_LDS
typedef __attribute__((address_space(3))) vi4 lds_vi4_t;
typedef __attribute__((address_space(1))) vi4 gbl_vi4_t;
__device__ __forceinline__ lds_vi4_t* as_lds(void* p) {
    // Low 32 bits of a generic LDS pointer are the LDS byte offset
    // (ISA: aperture in high bits).
    return (lds_vi4_t*)(unsigned int)(unsigned long long)p;
}
__device__ __forceinline__ gbl_vi4_t* as_gbl(const void* p) {
    return (gbl_vi4_t*)(unsigned long long)p;
}
__device__ __forceinline__ void wait_asynccnt0() {
#if __has_builtin(__builtin_amdgcn_s_wait_asynccnt)
    __builtin_amdgcn_s_wait_asynccnt(0);
#else
    asm volatile("s_wait_asynccnt 0x0" ::: "memory");
#endif
}
#endif

// ---- helpers ---------------------------------------------------------------

__device__ __forceinline__ unsigned short f2bf(float f) {
    unsigned int u = __float_as_uint(f);
    unsigned int r = u + 0x7FFFu + ((u >> 16) & 1u);   // round-to-nearest-even
    return (unsigned short)(r >> 16);
}

union AFrag { uint4 q[2]; v16bf v; };

// ---- kernel 0: zero output + expert counters -------------------------------

__global__ void moe_zero_kernel(float* out, int n, int* counts) {
    int i = blockIdx.x * blockDim.x + threadIdx.x;
    if (i < n)    out[i] = 0.0f;
    if (i < NEXP) counts[i] = 0;
}

// ---- kernel 1a: fp32 -> bf16 elementwise (activations) ---------------------

__global__ void moe_cvt_bf16_kernel(const float* __restrict__ in,
                                    unsigned short* __restrict__ out, int n) {
    int i = blockIdx.x * blockDim.x + threadIdx.x;
    if (i < n) out[i] = f2bf(in[i]);
}

// ---- kernel 1b: fp32 [E][K][N] -> bf16 [E][N][K] (weight transpose) --------
// 32x32 LDS-tiled transpose; block (32,8), grid (K/32, N/32, E).

__global__ void moe_cvt_transpose_kernel(const float* __restrict__ in,
                                         unsigned short* __restrict__ out) {
    __shared__ unsigned short tile[32][33];
    const int e  = blockIdx.z;
    const int k0 = blockIdx.x * 32;
    const int n0 = blockIdx.y * 32;
    const int tx = threadIdx.x;          // 0..31
    const int ty = threadIdx.y;          // 0..7
#pragma unroll
    for (int i = 0; i < 4; ++i) {
        int k = ty * 4 + i;
        tile[k][tx] = f2bf(in[((size_t)e * DIM + k0 + k) * DIM + n0 + tx]);
    }
    __syncthreads();
#pragma unroll
    for (int i = 0; i < 4; ++i) {
        int n = ty * 4 + i;
        out[((size_t)e * DIM + n0 + n) * DIM + k0 + tx] = tile[tx][n];
    }
}

// ---- kernel 2: gating (softmax over 8 experts, top-2, slot-weight quirk) ---

__global__ void moe_gate_kernel(const float* __restrict__ x,
                                const float* __restrict__ gw,
                                const float* __restrict__ gb,
                                int* __restrict__ counts,
                                int* __restrict__ rowlist,
                                float* __restrict__ wgtlist) {
    int t = blockIdx.x * blockDim.x + threadIdx.x;
    if (t >= TOK) return;

    float acc[NEXP];
#pragma unroll
    for (int e = 0; e < NEXP; ++e) acc[e] = gb[e];

    const float* xr = x + (size_t)t * DIM;
    for (int d = 0; d < DIM; ++d) {
        float xv = xr[d];
#pragma unroll
        for (int e = 0; e < NEXP; ++e) acc[e] += xv * gw[e * DIM + d];
    }

    float m = acc[0];
#pragma unroll
    for (int e = 1; e < NEXP; ++e) m = fmaxf(m, acc[e]);
    float p[NEXP], s = 0.0f;
#pragma unroll
    for (int e = 0; e < NEXP; ++e) { p[e] = __expf(acc[e] - m); s += p[e]; }
    float inv = 1.0f / s;
#pragma unroll
    for (int e = 0; e < NEXP; ++e) p[e] *= inv;

    // top-2 indices (ties -> lowest index, matching lax.top_k)
    int i0 = 0;
#pragma unroll
    for (int e = 1; e < NEXP; ++e) if (p[e] > p[i0]) i0 = e;
    int i1 = (i0 == 0) ? 1 : 0;
#pragma unroll
    for (int e = 0; e < NEXP; ++e)
        if (e != i0 && p[e] > p[i1]) i1 = e;

    // FAITHFUL quirk: combine weight of slot j is probs[..., j], j = 0,1.
    float w0 = p[0];
    float w1 = p[1];

    int pos0 = atomicAdd(&counts[i0], 1);
    rowlist[i0 * TOK + pos0] = t;
    wgtlist[i0 * TOK + pos0] = w0;

    int pos1 = atomicAdd(&counts[i1], 1);
    rowlist[i1 * TOK + pos1] = t;
    wgtlist[i1 * TOK + pos1] = w1;
}

// ---- kernel 3: tiled bf16 WMMA GEMM ----------------------------------------
// C[rows x 1024] = A[rows x 1024] * W^T, W stored transposed [N][K] bf16.
// Workgroup: 256 threads = 8 waves; block tile 128x64; BLK_K = 32.
// Wave (wm,wn): 32x32 tile -> 4 x v_wmma_f32_16x16x32_bf16 per K-step.
// mode 0: Hout[pos] = bf16(relu(acc + bias))          (stages 1 & 2)
// mode 1: Out[token] += wgt * (acc + bias)            (stage 3, scatter-add)

#define LDP 40   // padded LDS row stride in bf16 elements (80 B, 16B aligned)

__global__ void __launch_bounds__(256)
moe_gemm_kernel(const unsigned short* __restrict__ A,
                const unsigned short* __restrict__ BwT,   // [N][K] bf16
                const float* __restrict__ bias,
                const int* __restrict__ cnt,
                const int* __restrict__ rowlist,
                const float* __restrict__ wgtlist,
                unsigned short* __restrict__ Hout,
                float* __restrict__ Out,
                int gatherA, int mode) {
    const int rows = *cnt;
    const int m0 = blockIdx.x * 128;
    if (m0 >= rows) return;
    const int n0 = blockIdx.y * 64;

    __shared__ __align__(16) unsigned short As[128 * LDP]; // A rows [m][k]
    __shared__ __align__(16) unsigned short Bs[64 * LDP];  // W^T rows [n][k]

    const int tid  = threadIdx.x;
    const int lane = tid & 31;
    const int wave = tid >> 5;
    const int wm   = wave & 3;   // M sub-tile: rows [wm*32, +32)
    const int wn   = wave >> 2;  // N sub-tile: cols [wn*32, +32)
    const int half = lane >> 4;  // 0/1
    const int mr   = lane & 15;

    // A staging: 128 rows x 32 k bf16 = 512 uint4 -> 2 chunks per thread
    int ar[2], aq[2], arow[2]; bool avalid[2];
#pragma unroll
    for (int i = 0; i < 2; ++i) {
        int c = tid + i * 256;
        ar[i] = c >> 2;               // 0..127
        aq[i] = c & 3;                // 0..3
        int gpos = m0 + ar[i];
        avalid[i] = gpos < rows;
        arow[i] = avalid[i] ? (gatherA ? rowlist[gpos] : gpos) : 0;
    }
    // B staging: 64 rows x 32 k bf16 = 256 uint4 -> 1 chunk per thread
    const int bn = tid >> 2;          // 0..63
    const int bq = tid & 3;           // 0..3

    v8f acc00 = {}, acc01 = {}, acc10 = {}, acc11 = {};

    for (int k0 = 0; k0 < DIM; k0 += 32) {
        // ---- stage A tile ----
#pragma unroll
        for (int i = 0; i < 2; ++i) {
            const unsigned short* src = A + (size_t)arow[i] * DIM + k0 + aq[i] * 8;
            unsigned short*       dst = As + ar[i] * LDP + aq[i] * 8;
#ifdef HAS_ASYNC_LDS
            if (avalid[i]) {
                __builtin_amdgcn_global_load_async_to_lds_b128(
                    as_gbl(src), as_lds(dst), 0, 0);
            } else {
                *(uint4*)dst = make_uint4(0u, 0u, 0u, 0u);
            }
#else
            uint4 v = make_uint4(0u, 0u, 0u, 0u);
            if (avalid[i]) v = *(const uint4*)src;
            *(uint4*)dst = v;
#endif
        }
        // ---- stage B tile (already transposed in global: contiguous K) ----
        {
            const unsigned short* src = BwT + (size_t)(n0 + bn) * DIM + k0 + bq * 8;
            unsigned short*       dst = Bs + bn * LDP + bq * 8;
#ifdef HAS_ASYNC_LDS
            __builtin_amdgcn_global_load_async_to_lds_b128(
                as_gbl(src), as_lds(dst), 0, 0);
#else
            *(uint4*)dst = *(const uint4*)src;
#endif
        }
#ifdef HAS_ASYNC_LDS
        wait_asynccnt0();
#endif
        __syncthreads();

        // ---- A fragments (16x32 bf16): lane<16 -> K 0..7 & 16..23 of row mr,
        //      lane>=16 -> K 8..15 & 24..31 ----
        AFrag fa0, fa1;
        {
            int o0 = (wm * 32 + mr) * LDP + half * 8;
            fa0.q[0] = *(const uint4*)(As + o0);
            fa0.q[1] = *(const uint4*)(As + o0 + 16);
            int o1 = (wm * 32 + 16 + mr) * LDP + half * 8;
            fa1.q[0] = *(const uint4*)(As + o1);
            fa1.q[1] = *(const uint4*)(As + o1 + 16);
        }
        // ---- B fragments (32x16 bf16): lane<16 -> K 0..7 & 16..23 of col mr,
        //      lane>=16 -> K 8..15 & 24..31 (same order as A) ----
        AFrag fb0, fb1;
        {
            int o0 = (wn * 32 + mr) * LDP + half * 8;
            fb0.q[0] = *(const uint4*)(Bs + o0);
            fb0.q[1] = *(const uint4*)(Bs + o0 + 16);
            int o1 = (wn * 32 + 16 + mr) * LDP + half * 8;
            fb1.q[0] = *(const uint4*)(Bs + o1);
            fb1.q[1] = *(const uint4*)(Bs + o1 + 16);
        }

        acc00 = __builtin_amdgcn_wmma_f32_16x16x32_bf16(
                    false, fa0.v, false, fb0.v, (short)0, acc00, false, false);
        acc01 = __builtin_amdgcn_wmma_f32_16x16x32_bf16(
                    false, fa0.v, false, fb1.v, (short)0, acc01, false, false);
        acc10 = __builtin_amdgcn_wmma_f32_16x16x32_bf16(
                    false, fa1.v, false, fb0.v, (short)0, acc10, false, false);
        acc11 = __builtin_amdgcn_wmma_f32_16x16x32_bf16(
                    false, fa1.v, false, fb1.v, (short)0, acc11, false, false);

        __syncthreads();
    }

    // ---- epilogue ----
    // C layout: VGPR j, lane<16 -> (M=j, N=lane); lane>=16 -> (M=8+j, N=lane-16)
#pragma unroll
    for (int s = 0; s < 2; ++s) {
#pragma unroll
        for (int t = 0; t < 2; ++t) {
            const int col = n0 + wn * 32 + t * 16 + mr;
            const float bv = bias[col];
#pragma unroll
            for (int j = 0; j < 8; ++j) {
                int row  = wm * 32 + s * 16 + half * 8 + j;
                int gpos = m0 + row;
                if (gpos < rows) {
                    float a = (s == 0) ? ((t == 0) ? acc00[j] : acc01[j])
                                       : ((t == 0) ? acc10[j] : acc11[j]);
                    float v = a + bv;
                    if (mode == 0) {
                        v = fmaxf(v, 0.0f);
                        Hout[(size_t)gpos * DIM + col] = f2bf(v);
                    } else {
                        int   token = rowlist[gpos];
                        float w     = wgtlist[gpos];
                        Out[(size_t)token * DIM + col] += w * v; // unique per expert
                    }
                }
            }
        }
    }
}

// ---------------------------------------------------------------------------
// Host-side launcher
// ---------------------------------------------------------------------------

extern "C" void kernel_launch(void* const* d_in, const int* in_sizes, int n_in,
                              void* d_out, int out_size, void* d_ws, size_t ws_size,
                              hipStream_t stream) {
    const float* x      = (const float*)d_in[0];   // [T, D]
    const float* gate_w = (const float*)d_in[1];   // [E, D]
    const float* gate_b = (const float*)d_in[2];   // [E]
    const float* w1     = (const float*)d_in[3];   // [E, D, O]
    const float* b1     = (const float*)d_in[4];   // [E, O]
    const float* w2     = (const float*)d_in[5];   // [E, O, O]
    const float* b2     = (const float*)d_in[6];   // [E, O]
    const float* w3     = (const float*)d_in[7];   // [E, O, O]
    const float* b3     = (const float*)d_in[8];   // [E, O]
    float*       out    = (float*)d_out;           // [T, O]

    char* ws = (char*)d_ws;
    const size_t EW = (size_t)NEXP * DIM * DIM;    // elements per weight tensor

    size_t off = 0;
    unsigned short* xb  = (unsigned short*)(ws + off); off += (size_t)TOK * DIM * 2;
    unsigned short* w1t = (unsigned short*)(ws + off); off += EW * 2;
    unsigned short* w2t = (unsigned short*)(ws + off); off += EW * 2;
    unsigned short* w3t = (unsigned short*)(ws + off); off += EW * 2;
    unsigned short* h1  = (unsigned short*)(ws + off); off += (size_t)TOK * DIM * 2;
    unsigned short* h2  = (unsigned short*)(ws + off); off += (size_t)TOK * DIM * 2;
    int*   rowlist = (int*)  (ws + off); off += (size_t)NEXP * TOK * 4;
    float* wgtlist = (float*)(ws + off); off += (size_t)NEXP * TOK * 4;
    int*   counts  = (int*)  (ws + off); off += 64;
    (void)ws_size; (void)in_sizes; (void)n_in; (void)out_size;

    const int TPB = 256;

    // 0) zero output + counters (idempotent per call)
    moe_zero_kernel<<<(TOK * DIM + TPB - 1) / TPB, TPB, 0, stream>>>(
        out, TOK * DIM, counts);

    // 1) conversions: x elementwise; weights transposed to [E][N][K] bf16
    moe_cvt_bf16_kernel<<<(int)(((size_t)TOK * DIM + TPB - 1) / TPB), TPB, 0, stream>>>(
        x, xb, TOK * DIM);
    dim3 tgrid(DIM / 32, DIM / 32, NEXP), tblk(32, 8);
    moe_cvt_transpose_kernel<<<tgrid, tblk, 0, stream>>>(w1, w1t);
    moe_cvt_transpose_kernel<<<tgrid, tblk, 0, stream>>>(w2, w2t);
    moe_cvt_transpose_kernel<<<tgrid, tblk, 0, stream>>>(w3, w3t);

    // 2) gating + routing
    moe_gate_kernel<<<TOK / TPB, TPB, 0, stream>>>(
        x, gate_w, gate_b, counts, rowlist, wgtlist);

    // 3) per-expert sparse FFN chain (experts sequential -> no output atomics)
    dim3 grid(TOK / 128, DIM / 64);
    for (int e = 0; e < NEXP; ++e) {
        const size_t wo = (size_t)e * DIM * DIM;
        moe_gemm_kernel<<<grid, TPB, 0, stream>>>(
            xb, w1t + wo, b1 + e * DIM, counts + e,
            rowlist + e * TOK, wgtlist + e * TOK,
            h1, nullptr, /*gatherA=*/1, /*mode=*/0);
        moe_gemm_kernel<<<grid, TPB, 0, stream>>>(
            h1, w2t + wo, b2 + e * DIM, counts + e,
            rowlist + e * TOK, wgtlist + e * TOK,
            h2, nullptr, /*gatherA=*/0, /*mode=*/0);
        moe_gemm_kernel<<<grid, TPB, 0, stream>>>(
            h2, w3t + wo, b3 + e * DIM, counts + e,
            rowlist + e * TOK, wgtlist + e * TOK,
            nullptr, out, /*gatherA=*/0, /*mode=*/1);
    }
}
